// SlidingWindowAttention_64896955842747
// MI455X (gfx1250) — matrix operand order repaired and checked by reading, not compiled, not run on hardware
//
#include <hip/hip_runtime.h>
#include <hip/hip_bf16.h>

typedef _Float16 v16h __attribute__((ext_vector_type(16)));
typedef _Float16 v8h  __attribute__((ext_vector_type(8)));
typedef float    v8f  __attribute__((ext_vector_type(8)));

union H16 { v16h v; v8h h[2]; };

#define D_MODEL 1024
#define NHEAD   16
#define HD      64
#define SEQ     2048
#define WIN     2048
#define PSTRIDE 2080           // 16 x 2080 f32 band-score rows
#define ASTRIDE 2056           // 16 x 2056 f16 attn-weight rows (+8 pad -> 4-bank row skew)
#define LDS_S_BYTES   (16 * PSTRIDE * 4)            // 133120
#define LDS_A_BYTES   (16 * ASTRIDE * 2)            // 65792
#define LDS_P_BYTES   (4 * 256 * 4)                 // 4096
#define LDS_TOTAL     (LDS_S_BYTES + LDS_A_BYTES + LDS_P_BYTES)

__device__ __forceinline__ v8f wmma_f16(const H16& a, const H16& b, v8f c) {
  // D = A(16x32 f16) x B(32x16 f16) + C(16x16 f32)
  return __builtin_amdgcn_wmma_f32_16x16x32_f16(false, a.v, false, b.v,
                                                (short)0, c, false, false);
}

// ---------------- elementwise f32 -> f16 convert ----------------
__global__ void cvt_f16(const float* __restrict__ in, _Float16* __restrict__ out, int n) {
  for (int i = blockIdx.x * blockDim.x + threadIdx.x; i < n; i += gridDim.x * blockDim.x)
    out[i] = (_Float16)in[i];
}

// ---------------- qkv epilogue: scatter one 16x16 C block to q/k/vT ----------------
__device__ __forceinline__ void qkv_store8(float bias, const v8f& c, int mbase, int n, int hi,
                                           _Float16* __restrict__ qh,
                                           _Float16* __restrict__ kh,
                                           _Float16* __restrict__ vT) {
#pragma unroll
  for (int rr = 0; rr < 8; ++rr) {
    const int m = mbase + rr + (hi << 3);
    const float val = c[rr] + bias;
    if (n < D_MODEL) {                           // q : fold scale 1/8 (exact)
      qh[(size_t)(n >> 6) * (SEQ * HD) + (size_t)m * HD + (n & 63)] = (_Float16)(val * 0.125f);
    } else if (n < 2 * D_MODEL) {                // k
      const int nl = n - D_MODEL;
      kh[(size_t)(nl >> 6) * (SEQ * HD) + (size_t)m * HD + (nl & 63)] = (_Float16)val;
    } else {                                     // v, transposed per head: [H][HD][S]
      const int nl = n - 2 * D_MODEL;
      vT[(size_t)nl * SEQ + m] = (_Float16)val;
    }
  }
}

// ---------------- qkv = x @ w_qkv^T + b ; 32x32 register tile per wave ----------------
__global__ void qkv_gemm(const _Float16* __restrict__ xh,
                         const _Float16* __restrict__ wh,
                         const float*    __restrict__ bqkv,
                         _Float16* __restrict__ qh,
                         _Float16* __restrict__ kh,
                         _Float16* __restrict__ vT) {
  const int lane = threadIdx.x & 31;
  const int wave = threadIdx.x >> 5;
  const int n0   = (blockIdx.x * 8 + wave) * 32;
  const int m0   = blockIdx.y * 32;
  const int l15  = lane & 15;
  const int hi   = lane >> 4;
  const int kb   = hi * 8;
  const int kb2  = hi * 16;
  const _Float16* arow0 = xh + (size_t)(m0 + l15) * D_MODEL;
  const _Float16* arow1 = arow0 + (size_t)16 * D_MODEL;
  const _Float16* brow0 = wh + (size_t)(n0 + l15) * D_MODEL;
  const _Float16* brow1 = brow0 + (size_t)16 * D_MODEL;
  v8f c00 = {}, c01 = {}, c10 = {}, c11 = {};
  for (int kk = 0; kk < D_MODEL; kk += 32) {
    H16 a0, a1, b0, b1;
    a0.h[0] = *(const v8h*)(arow0 + kk + kb2);
    a0.h[1] = *(const v8h*)(arow0 + kk + kb2 + 8);
    a1.h[0] = *(const v8h*)(arow1 + kk + kb2);
    a1.h[1] = *(const v8h*)(arow1 + kk + kb2 + 8);
    b0.h[0] = *(const v8h*)(brow0 + kk + kb2);
    b0.h[1] = *(const v8h*)(brow0 + kk + kb2 + 8);
    b1.h[0] = *(const v8h*)(brow1 + kk + kb2);
    b1.h[1] = *(const v8h*)(brow1 + kk + kb2 + 8);
    c00 = wmma_f16(a0, b0, c00);
    c01 = wmma_f16(a0, b1, c01);
    c10 = wmma_f16(a1, b0, c10);
    c11 = wmma_f16(a1, b1, c11);
  }
  const int nA = n0 + l15, nB = n0 + 16 + l15;
  const float biasA = bqkv[nA], biasB = bqkv[nB];
  qkv_store8(biasA, c00, m0,      nA, hi, qh, kh, vT);
  qkv_store8(biasB, c01, m0,      nB, hi, qh, kh, vT);
  qkv_store8(biasA, c10, m0 + 16, nA, hi, qh, kh, vT);
  qkv_store8(biasB, c11, m0 + 16, nB, hi, qh, kh, vT);
}

// ---------------- fused band-scores + softmax + attn@v, one WG per (head, 16-row block) ----------------
__global__ void swa_attn(const _Float16* __restrict__ qh,
                         const _Float16* __restrict__ kh,
                         const _Float16* __restrict__ vT,
                         float*    __restrict__ attn_g,  // [H][S][W] f32 output
                         _Float16* __restrict__ aoh) {   // [S][D] f16 attention output
  extern __shared__ char smem[];
  float*    S    = (float*)smem;                         // 16 x PSTRIDE f32 band scores
  _Float16* AT   = (_Float16*)(smem + LDS_S_BYTES);      // 16 x ASTRIDE f16 attn weights
  float*    PART = (float*)(smem + LDS_S_BYTES + LDS_A_BYTES); // 4 x 16 x 16 f32 partials

  const int lane = threadIdx.x & 31;
  const int wave = threadIdx.x >> 5;
  const int head = blockIdx.x >> 7;
  const int i0   = (blockIdx.x & 127) * 16;
  const int l15  = lane & 15;
  const int hi   = lane >> 4;
  const int kb   = hi * 8;
  const int kb2  = hi * 16;
  const int t_start = i0 - WIN / 2;

  // ---- Phase A: P[r][p] = q_{i0+r} . k[t_start + p], p in [0, 2080), zero when key OOB
  const _Float16* qrow = qh + ((size_t)head * SEQ + i0 + l15) * HD;
  H16 a0, a1;
  a0.h[0] = *(const v8h*)(qrow + kb2);
  a0.h[1] = *(const v8h*)(qrow + kb2 + 8);
  a1.h[0] = *(const v8h*)(qrow + 32 + kb2);
  a1.h[1] = *(const v8h*)(qrow + 32 + kb2 + 8);
  const v8h z = {};
  for (int tile = wave; tile < 130; tile += 8) {
    const int p0 = tile * 16;
    const int tn = t_start + p0 + l15;
    const bool valid = (tn >= 0) && (tn < SEQ);
    const int tc = valid ? tn : 0;
    const _Float16* krow = kh + ((size_t)head * SEQ + tc) * HD;
    H16 b0, b1;
    b0.h[0] = valid ? *(const v8h*)(krow + kb2)          : z;
    b0.h[1] = valid ? *(const v8h*)(krow + kb2 + 8)      : z;
    b1.h[0] = valid ? *(const v8h*)(krow + 32 + kb2)     : z;
    b1.h[1] = valid ? *(const v8h*)(krow + 32 + kb2 + 8) : z;
    v8f c = {};
    c = wmma_f16(a0, b0, c);
    c = wmma_f16(a1, b1, c);
#pragma unroll
    for (int rr = 0; rr < 8; ++rr)
      S[(rr + (hi << 3)) * PSTRIDE + p0 + l15] = c[rr];
  }
  __syncthreads();

  // ---- Phase B: softmax over all 2048 window cols (zeros included, as in reference);
  //      write attn f32 to global and f16 weights to LDS.
  for (int rr = 0; rr < 2; ++rr) {
    const int r = wave + rr * 8;
    const float* Srow = S + r * PSTRIDE;              // row r's window lives at Srow[r + j]
    _Float16* Arow = AT + r * ASTRIDE;
    float mx = -1e30f;
    for (int j = lane; j < WIN; j += 32) mx = fmaxf(mx, Srow[r + j]);
#pragma unroll
    for (int o = 16; o > 0; o >>= 1) mx = fmaxf(mx, __shfl_xor(mx, o, 32));
    float sum = 0.f;
    for (int j = lane; j < WIN; j += 32) sum += __expf(Srow[r + j] - mx);
#pragma unroll
    for (int o = 16; o > 0; o >>= 1) sum += __shfl_xor(sum, o, 32);
    const float inv = 1.0f / sum;                     // sum >= exp(-mx) > 0 : no NaN possible
    float* dst = attn_g + ((size_t)(head * SEQ) + i0 + r) * WIN;
    for (int j = lane; j < WIN; j += 32) {
      const float av = __expf(Srow[r + j] - mx) * inv;
      dst[j] = av;
      Arow[j] = (_Float16)av;
    }
  }
  __syncthreads();

  // ---- Phase C: out16 = attn(16x2048) @ v[j] (column-indexed, faithful to source).
  //      8 waves = 4 hd-slices x 2 j-halves; LDS reduction of the j-halves.
  const int ntile = wave & 3;
  const int jhalf = wave >> 2;
  const _Float16* ATrow = AT + l15 * ASTRIDE;                                 // A row = lane&15
  const _Float16* vrow  = vT + ((size_t)head * HD + ntile * 16 + l15) * SEQ;  // B col = lane&15
  v8f c = {};
  const int jb0 = jhalf * (WIN / 2);
  for (int jb = jb0; jb < jb0 + WIN / 2; jb += 32) {
    H16 a, b;
    a.h[0] = *(const v8h*)(ATrow + jb + kb2);
    a.h[1] = *(const v8h*)(ATrow + jb + kb2 + 8);
    b.h[0] = *(const v8h*)(vrow + jb + kb2);
    b.h[1] = *(const v8h*)(vrow + jb + kb2 + 8);
    c = wmma_f16(a, b, c);
  }
  float* part = PART + ntile * 256;
  if (jhalf == 1) {
#pragma unroll
    for (int rr = 0; rr < 8; ++rr)
      part[(rr + (hi << 3)) * 16 + l15] = c[rr];
  }
  __syncthreads();
  if (jhalf == 0) {
#pragma unroll
    for (int rr = 0; rr < 8; ++rr) {
      const int mr = rr + (hi << 3);
      const float val = c[rr] + part[mr * 16 + l15];
      aoh[(size_t)(i0 + mr) * D_MODEL + head * HD + ntile * 16 + l15] = (_Float16)val;
    }
  }
}

// ---------------- out = attn_out @ w_proj^T + b_proj ; 32x32 register tile per wave ----------------
__global__ void proj_gemm(const _Float16* __restrict__ aoh,
                          const _Float16* __restrict__ wph,
                          const float*    __restrict__ bp,
                          float* __restrict__ out) {
  const int lane = threadIdx.x & 31;
  const int wave = threadIdx.x >> 5;
  const int n0   = (blockIdx.x * 8 + wave) * 32;
  const int m0   = blockIdx.y * 32;
  const int l15  = lane & 15;
  const int hi   = lane >> 4;
  const int kb   = hi * 8;
  const int kb2  = hi * 16;
  const _Float16* arow0 = aoh + (size_t)(m0 + l15) * D_MODEL;
  const _Float16* arow1 = arow0 + (size_t)16 * D_MODEL;
  const _Float16* brow0 = wph + (size_t)(n0 + l15) * D_MODEL;
  const _Float16* brow1 = brow0 + (size_t)16 * D_MODEL;
  v8f c00 = {}, c01 = {}, c10 = {}, c11 = {};
  for (int kk = 0; kk < D_MODEL; kk += 32) {
    H16 a0, a1, b0, b1;
    a0.h[0] = *(const v8h*)(arow0 + kk + kb2);
    a0.h[1] = *(const v8h*)(arow0 + kk + kb2 + 8);
    a1.h[0] = *(const v8h*)(arow1 + kk + kb2);
    a1.h[1] = *(const v8h*)(arow1 + kk + kb2 + 8);
    b0.h[0] = *(const v8h*)(brow0 + kk + kb2);
    b0.h[1] = *(const v8h*)(brow0 + kk + kb2 + 8);
    b1.h[0] = *(const v8h*)(brow1 + kk + kb2);
    b1.h[1] = *(const v8h*)(brow1 + kk + kb2 + 8);
    c00 = wmma_f16(a0, b0, c00);
    c01 = wmma_f16(a0, b1, c01);
    c10 = wmma_f16(a1, b0, c10);
    c11 = wmma_f16(a1, b1, c11);
  }
  const int nA = n0 + l15, nB = n0 + 16 + l15;
  const float biasA = bp[nA], biasB = bp[nB];
#pragma unroll
  for (int rr = 0; rr < 8; ++rr) {
    const int mA = m0 + rr + (hi << 3);
    out[(size_t)mA * D_MODEL + nA]        = c00[rr] + biasA;
    out[(size_t)mA * D_MODEL + nB]        = c01[rr] + biasB;
    out[(size_t)(mA + 16) * D_MODEL + nA] = c10[rr] + biasA;
    out[(size_t)(mA + 16) * D_MODEL + nB] = c11[rr] + biasB;
  }
}

extern "C" void kernel_launch(void* const* d_in, const int* in_sizes, int n_in,
                              void* d_out, int out_size, void* d_ws, size_t ws_size,
                              hipStream_t stream) {
  const float* x    = (const float*)d_in[0];
  const float* wqkv = (const float*)d_in[1];
  const float* bqkv = (const float*)d_in[2];
  const float* wp   = (const float*)d_in[3];
  const float* bp   = (const float*)d_in[4];
  float* out = (float*)d_out;
  char* ws = (char*)d_ws;

  // workspace layout (bytes)
  _Float16* xh  = (_Float16*)(ws + 0);          // 2048x1024 f16  (4 MB)
  _Float16* wqh = (_Float16*)(ws + 4194304);    // 3072x1024 f16  (6 MB)
  _Float16* wph = (_Float16*)(ws + 10485760);   // 1024x1024 f16  (2 MB)
  _Float16* qh  = (_Float16*)(ws + 12582912);   // [16][2048][64] f16 (4 MB)
  _Float16* kh  = (_Float16*)(ws + 16777216);   // [16][2048][64] f16 (4 MB)
  _Float16* vT  = (_Float16*)(ws + 20971520);   // [16][64][2048] f16 (4 MB)
  _Float16* aoh = (_Float16*)(ws + 25165824);   // [2048][1024]  f16 (4 MB)

  cvt_f16<<<(2097152 + 255) / 256, 256, 0, stream>>>(x,    xh,  2097152);
  cvt_f16<<<(3145728 + 255) / 256, 256, 0, stream>>>(wqkv, wqh, 3145728);
  cvt_f16<<<(1048576 + 255) / 256, 256, 0, stream>>>(wp,   wph, 1048576);

  // qkv GEMM: 32x32 tiles -> 64(m) x 96(n) tiles, 8 n-tiles per 256-thread block
  qkv_gemm<<<dim3(12, 64), 256, 0, stream>>>(xh, wqh, bqkv, qh, kh, vT);

  // fused attention: one WG per (head, 16-row block); ~198 KB dynamic LDS
  float* attn_g = out + (size_t)SEQ * D_MODEL;  // outputs: [out | attn] concatenated
  swa_attn<<<NHEAD * (SEQ / 16), 256, LDS_TOTAL, stream>>>(qh, kh, vT, attn_g, aoh);

  // projection GEMM: 32x32 tiles -> 64(m) x 32(n) tiles
  proj_gemm<<<dim3(4, 64), 256, 0, stream>>>(aoh, wph, bp, out);
}
